// SmolLM2MLP_1425929142352
// MI455X (gfx1250) — matrix operand order repaired and checked by reading, not compiled, not run on hardware
//
#include <hip/hip_runtime.h>
#include <hip/hip_bf16.h>

typedef __attribute__((ext_vector_type(16))) _Float16 v16h;
typedef __attribute__((ext_vector_type(8)))  _Float16 v8h;
typedef __attribute__((ext_vector_type(8)))  float    v8f;

constexpr int T = 4096;   // B*S tokens
constexpr int H = 1024;
constexpr int I = 2048;
constexpr int E = 7;      // routed experts

// ---------------- workspace layout (bytes) ----------------
constexpr size_t SZ_XF16 = (size_t)T * H * 2;
constexpr size_t SZ_W    = (size_t)H * I * 2;          // one f16 weight matrix
constexpr size_t OFF_XF16 = 0;
constexpr size_t OFF_WSG  = OFF_XF16 + SZ_XF16;        // shared gate Wt[I][H]
constexpr size_t OFF_WSU  = OFF_WSG + SZ_W;            // shared up   Wt[I][H]
constexpr size_t OFF_WSD  = OFF_WSU + SZ_W;            // shared down Wt[H][I]
constexpr size_t OFF_WRG  = OFF_WSD + SZ_W;            // routed gate E x Wt[I][H]
constexpr size_t OFF_WRU  = OFF_WRG + (size_t)E * SZ_W;
constexpr size_t OFF_WRD  = OFF_WRU + (size_t)E * SZ_W;
constexpr size_t OFF_META = OFF_WRD + (size_t)E * SZ_W; // 32 ints: [0..6]=counts [8..14]=fill [16..23]=offsets
constexpr size_t OFF_TOKE = OFF_META + 128;             // int [2T] expert id per (token,k)
constexpr size_t OFF_TOKW = OFF_TOKE + (size_t)2 * T * 4; // float[2T] normalized score
constexpr size_t OFF_TOKS = OFF_TOKW + (size_t)2 * T * 4; // int  [2T] slot per (token,k)
constexpr size_t OFF_STOK = OFF_TOKS + (size_t)2 * T * 4; // int  [2T] token per slot
constexpr size_t OFF_HSH  = (OFF_STOK + (size_t)2 * T * 4 + 255) & ~(size_t)255; // f16 [T][I]
constexpr size_t OFF_HRT  = OFF_HSH + (size_t)T * I * 2;   // f16 [2T][I] compacted
constexpr size_t OFF_ROUT = OFF_HRT + (size_t)2 * T * I * 2; // f32 [2T][H] compacted

// ---------------- helper kernels ----------------
__global__ void k_cvt(const float* __restrict__ src, _Float16* __restrict__ dst, int n) {
  int i = blockIdx.x * blockDim.x + threadIdx.x;
  if (i < n) dst[i] = (_Float16)src[i];
}

// Tiled transpose-convert: src [b][R][C] fp32 (row-major) -> dst [b][C][R] f16.
// 32x32 tile through LDS so both global reads and writes are coalesced.
__global__ __launch_bounds__(256) void k_cvt_tr(const float* __restrict__ src,
                                                _Float16* __restrict__ dst, int R, int C) {
  __shared__ _Float16 tile[32][40];  // +8 pad to dodge bank conflicts
  size_t b = blockIdx.z;
  const float* s = src + b * (size_t)R * C;
  _Float16* d = dst + b * (size_t)R * C;
  int c0 = blockIdx.x * 32, r0 = blockIdx.y * 32;
  int tx = threadIdx.x & 31, ty = threadIdx.x >> 5;  // 32 x 8 threads
#pragma unroll
  for (int j = 0; j < 4; ++j)
    tile[ty + j * 8][tx] = (_Float16)s[(size_t)(r0 + ty + j * 8) * C + c0 + tx];
  __syncthreads();
#pragma unroll
  for (int j = 0; j < 4; ++j)
    d[(size_t)(c0 + ty + j * 8) * R + r0 + tx] = tile[tx][ty + j * 8];
}

__global__ void k_init(int* meta) {
  if (threadIdx.x < 32) meta[threadIdx.x] = 0;
}

__global__ void k_router(const float* __restrict__ x, const float* __restrict__ rw,
                         const float* __restrict__ rb, int* meta,
                         int* __restrict__ tokE, float* __restrict__ tokW) {
  int t = blockIdx.x * blockDim.x + threadIdx.x;
  if (t >= T) return;
  const float* xr = x + (size_t)t * H;
  float logit[E];
#pragma unroll
  for (int e = 0; e < E; ++e) logit[e] = rb[e];
  for (int h = 0; h < H; ++h) {
    float xv = xr[h];
    const float* w = rw + (size_t)h * E;
#pragma unroll
    for (int e = 0; e < E; ++e) logit[e] += xv * w[e];
  }
  float p[E];
#pragma unroll
  for (int e = 0; e < E; ++e) p[e] = 1.0f / (1.0f + __expf(-logit[e]));
  // top-2, ties -> lowest index (matches lax.top_k)
  int i0 = 0; float s0 = p[0];
#pragma unroll
  for (int e = 1; e < E; ++e) if (p[e] > s0) { s0 = p[e]; i0 = e; }
  int i1 = -1; float s1 = -1e30f;
#pragma unroll
  for (int e = 0; e < E; ++e) if (e != i0 && p[e] > s1) { s1 = p[e]; i1 = e; }
  float inv = 1.0f / (s0 + s1);
  tokE[2 * t + 0] = i0; tokW[2 * t + 0] = s0 * inv;
  tokE[2 * t + 1] = i1; tokW[2 * t + 1] = s1 * inv;
  atomicAdd(&meta[i0], 1);
  atomicAdd(&meta[i1], 1);
}

__global__ void k_scan(int* meta) {
  if (threadIdx.x == 0) {
    int off = 0;
    for (int e = 0; e < E; ++e) { meta[16 + e] = off; off += meta[e]; }
    meta[16 + E] = off;
  }
}

__global__ void k_scatter(const int* __restrict__ tokE, int* meta,
                          int* __restrict__ slotTok, int* __restrict__ tokSlot) {
  int t = blockIdx.x * blockDim.x + threadIdx.x;
  if (t >= T) return;
#pragma unroll
  for (int k = 0; k < 2; ++k) {
    int e = tokE[2 * t + k];
    int p = atomicAdd(&meta[8 + e], 1);
    int slot = meta[16 + e] + p;
    slotTok[slot] = t;
    tokSlot[2 * t + k] = slot;
  }
}

// ---------------- stage 1: hidden = silu(x@G) * (x@U), gathered per expert ----------------
__global__ __launch_bounds__(256) void k_stage1(
    const _Float16* __restrict__ xf16,
    const _Float16* __restrict__ wsg, const _Float16* __restrict__ wsu,
    const _Float16* __restrict__ wrg, const _Float16* __restrict__ wru,
    const int* __restrict__ meta, const int* __restrict__ slotTok,
    _Float16* __restrict__ Hsh, _Float16* __restrict__ Hrt) {
  __shared__ __align__(64) _Float16 As[2][128][32];
  __shared__ __align__(64) _Float16 Bs[2][2][64][32];

  const int e = blockIdx.z;
  int cnt, base;
  const _Float16 *WG, *WU;
  _Float16* Hdst;
  if (e < E) {
    cnt = meta[e]; base = meta[16 + e];
    WG = wrg + (size_t)e * H * I;
    WU = wru + (size_t)e * H * I;
    Hdst = Hrt + (size_t)base * I;
  } else {
    cnt = T; base = 0;
    WG = wsg; WU = wsu; Hdst = Hsh;
  }
  const int mtile = blockIdx.x;
  if (mtile * 128 >= cnt) return;
  const int nbase = blockIdx.y * 64;

  const int tid = threadIdx.x;
  const int lane = tid & 31;
  const int wave = tid >> 5;
  const int wm = (wave & 3) * 32;   // 4 waves over M
  const int wn = (wave >> 2) * 32;  // 2 waves over N

  // A load assignment: 2 threads per row, 16 halfs (32B) each
  const int rA = tid >> 1, segA = tid & 1;
  const int mrow = mtile * 128 + rA;
  const _Float16* arow;
  if (mrow < cnt) {
    int tok = (e < E) ? slotTok[base + mrow] : mrow;
    arow = xf16 + (size_t)tok * H;
  } else {
    arow = xf16;  // dummy row, results discarded
  }
  // B load assignment: lower 128 threads -> gate, upper -> up; 2 threads per n-row
  const int matB = tid >> 7;
  const int qB = tid & 127;
  const int nB = qB >> 1, segB = qB & 1;
  const _Float16* brow = ((matB == 0) ? WG : WU) + (size_t)(nbase + nB) * H;

  v8f zero = {0.f, 0.f, 0.f, 0.f, 0.f, 0.f, 0.f, 0.f};
  v8f acc[2][2][2];  // [gate/up][mt][nt]
#pragma unroll
  for (int a = 0; a < 2; ++a)
#pragma unroll
    for (int b = 0; b < 2; ++b)
#pragma unroll
      for (int c = 0; c < 2; ++c) acc[a][b][c] = zero;

  uint4 av0, av1, bv0, bv1;
#define S1_LOAD_G(kk)                                              \
  do {                                                             \
    const uint4* asrc = (const uint4*)(arow + (kk) + segA * 16);   \
    av0 = asrc[0]; av1 = asrc[1];                                  \
    const uint4* bsrc = (const uint4*)(brow + (kk) + segB * 16);   \
    bv0 = bsrc[0]; bv1 = bsrc[1];                                  \
  } while (0)
#define S1_STORE_L(p)                                              \
  do {                                                             \
    *(uint4*)&As[p][rA][segA * 16] = av0;                          \
    *(uint4*)&As[p][rA][segA * 16 + 8] = av1;                      \
    *(uint4*)&Bs[p][matB][nB][segB * 16] = bv0;                    \
    *(uint4*)&Bs[p][matB][nB][segB * 16 + 8] = bv1;                \
  } while (0)

  constexpr int NS = H / 32;
  S1_LOAD_G(0);
  S1_STORE_L(0);
  const int arf = lane & 15, kh = lane >> 4;
  for (int step = 0; step < NS; ++step) {
    const int p = step & 1;
    const bool more = (step + 1) < NS;
    if (more) S1_LOAD_G((step + 1) * 32);  // next tile in flight during compute
    __syncthreads();                        // buf[p] writes visible

    v16h af[2];
#pragma unroll
    for (int mt = 0; mt < 2; ++mt) {
      v8h lo = *(const v8h*)&As[p][wm + mt * 16 + arf][kh * 16];
      v8h hi = *(const v8h*)&As[p][wm + mt * 16 + arf][kh * 16 + 8];
#pragma unroll
      for (int i2 = 0; i2 < 8; ++i2) { af[mt][i2] = lo[i2]; af[mt][8 + i2] = hi[i2]; }
    }
#pragma unroll
    for (int mat = 0; mat < 2; ++mat)
#pragma unroll
      for (int nt = 0; nt < 2; ++nt) {
        v16h bf = *(const v16h*)&Bs[p][mat][wn + nt * 16 + arf][kh * 16];
#pragma unroll
        for (int mt = 0; mt < 2; ++mt)
          acc[mat][mt][nt] = __builtin_amdgcn_wmma_f32_16x16x32_f16(
              false, af[mt], false, bf, (short)0, acc[mat][mt][nt], false, false);
      }
    if (more) S1_STORE_L(p ^ 1);  // all waves finished reading p^1 (barrier above)
  }
#undef S1_LOAD_G
#undef S1_STORE_L

  // epilogue: silu(gate) * up -> f16 hidden (fast rcp, not IEEE divide)
#pragma unroll
  for (int mt = 0; mt < 2; ++mt)
#pragma unroll
    for (int nt = 0; nt < 2; ++nt)
#pragma unroll
      for (int r = 0; r < 8; ++r) {
        int m = mtile * 128 + wm + mt * 16 + (lane >> 4) * 8 + r;
        if (m < cnt) {
          int c = nbase + wn + nt * 16 + (lane & 15);
          float g = acc[0][mt][nt][r];
          float u = acc[1][mt][nt][r];
          float s = g * __builtin_amdgcn_rcpf(1.0f + __expf(-g));
          Hdst[(size_t)m * I + c] = (_Float16)(s * u);
        }
      }
}

// ---------------- stage 2: down projection ----------------
__global__ __launch_bounds__(256) void k_stage2(
    const _Float16* __restrict__ Hsh, const _Float16* __restrict__ Hrt,
    const _Float16* __restrict__ wsd, const _Float16* __restrict__ wrd,
    const int* __restrict__ meta,
    float* __restrict__ rOut, float* __restrict__ dOut) {
  __shared__ __align__(64) _Float16 As[2][128][32];
  __shared__ __align__(64) _Float16 Bs[2][64][32];

  const int e = blockIdx.z;
  int cnt, base;
  const _Float16 *WD, *Hsrc;
  if (e < E) {
    cnt = meta[e]; base = meta[16 + e];
    WD = wrd + (size_t)e * H * I;
    Hsrc = Hrt + (size_t)base * I;
  } else {
    cnt = T; base = 0;
    WD = wsd; Hsrc = Hsh;
  }
  const int mtile = blockIdx.x;
  if (mtile * 128 >= cnt) return;
  const int nbase = blockIdx.y * 64;

  const int tid = threadIdx.x;
  const int lane = tid & 31;
  const int wave = tid >> 5;
  const int wm = (wave & 3) * 32;
  const int wn = (wave >> 2) * 32;

  const int rA = tid >> 1, segA = tid & 1;
  const int mrow = mtile * 128 + rA;
  const _Float16* arow = Hsrc + (size_t)((mrow < cnt) ? mrow : 0) * I;
  const int nB = tid >> 2, segB = tid & 3;  // 4 threads per n-row, 8 halfs each
  const _Float16* brow = WD + (size_t)(nbase + nB) * I;

  v8f zero = {0.f, 0.f, 0.f, 0.f, 0.f, 0.f, 0.f, 0.f};
  v8f acc[2][2];
#pragma unroll
  for (int a = 0; a < 2; ++a)
#pragma unroll
    for (int b = 0; b < 2; ++b) acc[a][b] = zero;

  uint4 av0, av1, bv;
#define S2_LOAD_G(kk)                                             \
  do {                                                            \
    const uint4* asrc = (const uint4*)(arow + (kk) + segA * 16);  \
    av0 = asrc[0]; av1 = asrc[1];                                 \
    bv = *(const uint4*)(brow + (kk) + segB * 8);                 \
  } while (0)
#define S2_STORE_L(p)                                             \
  do {                                                            \
    *(uint4*)&As[p][rA][segA * 16] = av0;                         \
    *(uint4*)&As[p][rA][segA * 16 + 8] = av1;                     \
    *(uint4*)&Bs[p][nB][segB * 8] = bv;                           \
  } while (0)

  constexpr int NS = I / 32;
  S2_LOAD_G(0);
  S2_STORE_L(0);
  const int arf = lane & 15, kh = lane >> 4;
  for (int step = 0; step < NS; ++step) {
    const int p = step & 1;
    const bool more = (step + 1) < NS;
    if (more) S2_LOAD_G((step + 1) * 32);
    __syncthreads();

    v16h af[2];
#pragma unroll
    for (int mt = 0; mt < 2; ++mt) {
      v8h lo = *(const v8h*)&As[p][wm + mt * 16 + arf][kh * 16];
      v8h hi = *(const v8h*)&As[p][wm + mt * 16 + arf][kh * 16 + 8];
#pragma unroll
      for (int i2 = 0; i2 < 8; ++i2) { af[mt][i2] = lo[i2]; af[mt][8 + i2] = hi[i2]; }
    }
#pragma unroll
    for (int nt = 0; nt < 2; ++nt) {
      v16h bf = *(const v16h*)&Bs[p][wn + nt * 16 + arf][kh * 16];
#pragma unroll
      for (int mt = 0; mt < 2; ++mt)
        acc[mt][nt] = __builtin_amdgcn_wmma_f32_16x16x32_f16(
            false, af[mt], false, bf, (short)0, acc[mt][nt], false, false);
    }
    if (more) S2_STORE_L(p ^ 1);
  }
#undef S2_LOAD_G
#undef S2_STORE_L

#pragma unroll
  for (int mt = 0; mt < 2; ++mt)
#pragma unroll
    for (int nt = 0; nt < 2; ++nt)
#pragma unroll
      for (int r = 0; r < 8; ++r) {
        int m = mtile * 128 + wm + mt * 16 + (lane >> 4) * 8 + r;
        if (m < cnt) {
          int c = nbase + wn + nt * 16 + (lane & 15);
          float v = acc[mt][nt][r];
          if (e < E) rOut[(size_t)(base + m) * H + c] = v;
          else       dOut[(size_t)m * H + c] = v;  // shared expert writes full output
        }
      }
}

// ---------------- combine: out += w0*routed(slot0) + w1*routed(slot1) ----------------
__global__ void k_combine(const float* __restrict__ rOut,
                          const int* __restrict__ tokSlot,
                          const float* __restrict__ tokW,
                          float* __restrict__ out) {
  size_t i = (size_t)blockIdx.x * blockDim.x + threadIdx.x;
  if (i >= (size_t)T * H) return;
  int t = (int)(i / H), h = (int)(i % H);
  int s0 = tokSlot[2 * t], s1 = tokSlot[2 * t + 1];
  float w0 = tokW[2 * t], w1 = tokW[2 * t + 1];
  out[i] += w0 * rOut[(size_t)s0 * H + h] + w1 * rOut[(size_t)s1 * H + h];
}

// ---------------- launch ----------------
extern "C" void kernel_launch(void* const* d_in, const int* in_sizes, int n_in,
                              void* d_out, int out_size, void* d_ws, size_t ws_size,
                              hipStream_t stream) {
  (void)in_sizes; (void)n_in; (void)out_size; (void)ws_size;
  const float* x   = (const float*)d_in[0];
  const float* sgw = (const float*)d_in[1];
  const float* suw = (const float*)d_in[2];
  const float* sdw = (const float*)d_in[3];
  const float* rgw = (const float*)d_in[4];
  const float* ruw = (const float*)d_in[5];
  const float* rdw = (const float*)d_in[6];
  const float* rw  = (const float*)d_in[7];
  const float* rb  = (const float*)d_in[8];
  float* out = (float*)d_out;

  char* ws = (char*)d_ws;
  _Float16* xf16 = (_Float16*)(ws + OFF_XF16);
  _Float16* wsg  = (_Float16*)(ws + OFF_WSG);
  _Float16* wsu  = (_Float16*)(ws + OFF_WSU);
  _Float16* wsd  = (_Float16*)(ws + OFF_WSD);
  _Float16* wrg  = (_Float16*)(ws + OFF_WRG);
  _Float16* wru  = (_Float16*)(ws + OFF_WRU);
  _Float16* wrd  = (_Float16*)(ws + OFF_WRD);
  int*      meta = (int*)(ws + OFF_META);
  int*      tokE = (int*)(ws + OFF_TOKE);
  float*    tokW = (float*)(ws + OFF_TOKW);
  int*   tokSlot = (int*)(ws + OFF_TOKS);
  int*   slotTok = (int*)(ws + OFF_STOK);
  _Float16* Hsh  = (_Float16*)(ws + OFF_HSH);
  _Float16* Hrt  = (_Float16*)(ws + OFF_HRT);
  float*    rOut = (float*)(ws + OFF_ROUT);

  k_cvt<<<(T * H) / 256, 256, 0, stream>>>(x, xf16, T * H);
  // weight convert+transpose (coalesced 32x32 LDS tiles)
  k_cvt_tr<<<dim3(I / 32, H / 32, 1), 256, 0, stream>>>(sgw, wsg, H, I);
  k_cvt_tr<<<dim3(I / 32, H / 32, 1), 256, 0, stream>>>(suw, wsu, H, I);
  k_cvt_tr<<<dim3(H / 32, I / 32, 1), 256, 0, stream>>>(sdw, wsd, I, H);
  k_cvt_tr<<<dim3(I / 32, H / 32, E), 256, 0, stream>>>(rgw, wrg, H, I);
  k_cvt_tr<<<dim3(I / 32, H / 32, E), 256, 0, stream>>>(ruw, wru, H, I);
  k_cvt_tr<<<dim3(H / 32, I / 32, E), 256, 0, stream>>>(rdw, wrd, I, H);
  k_init<<<1, 32, 0, stream>>>(meta);
  k_router<<<T / 256, 256, 0, stream>>>(x, rw, rb, meta, tokE, tokW);
  k_scan<<<1, 1, 0, stream>>>(meta);
  k_scatter<<<T / 256, 256, 0, stream>>>(tokE, meta, slotTok, tokSlot);
  k_stage1<<<dim3(T / 128, I / 64, E + 1), 256, 0, stream>>>(
      xf16, wsg, wsu, wrg, wru, meta, slotTok, Hsh, Hrt);
  k_stage2<<<dim3(T / 128, H / 64, E + 1), 256, 0, stream>>>(
      Hsh, Hrt, wsd, wrd, meta, rOut, out);
  k_combine<<<(T * H) / 256, 256, 0, stream>>>(rOut, tokSlot, tokW, out);
}
